// Glue_50637664419855
// MI455X (gfx1250) — matrix operand order repaired and checked by reading, not compiled, not run on hardware
//
#include <hip/hip_runtime.h>
#include <hip/hip_fp16.h>

// ---------------------------------------------------------------------------
// MI455X (gfx1250) implementation. All dense GEMMs (projections, MLPs,
// attention QK^T / P*V, score matrix) run through V_WMMA_F32_16X16X32_F16
// (f16 inputs, f32 accumulate). GEMM waves compute a compile-time 32x(NT*16)
// tile (2 A-frags x NT B-frags -> 2*NT WMMAs per k-step) with a straight-line
// inner loop (no divergence -> no accumulator spills). Sinkhorn (67MB) is
// L2-resident streaming (192MB L2). A/B fragments use contiguous k (slot i <- k = 16*khalf + i).
// ---------------------------------------------------------------------------

typedef _Float16 h16;
typedef __attribute__((ext_vector_type(16))) _Float16 v16h;
typedef __attribute__((ext_vector_type(8)))  _Float16 v8h;
typedef __attribute__((ext_vector_type(8)))  float    v8f;

#define NROWS 8192
#define NHALF 4096
#define EMB   256
#define KEYD  32
#define VALD  64
#define CATD  320
#define NP    4097
#define NPSQ  (4097LL*4097LL)

__device__ inline v8f wmma_f16f32(v16h a, v16h b, v8f c) {
  return __builtin_amdgcn_wmma_f32_16x16x32_f16(false, a, false, b, (short)0, c,
                                                false, false);
}

__device__ inline v16h load_afrag(const h16* __restrict__ arow, int k0, int khalf) {
  // A fragment (16x32 f16): slot i <- k0+khalf*16+i (contiguous)
  v8h lo = *(const v8h*)(arow + k0 + khalf * 16);
  v8h hi = *(const v8h*)(arow + k0 + khalf * 16 + 8);
  v16h a;
#pragma unroll
  for (int i = 0; i < 8; ++i) { a[i] = lo[i]; a[8 + i] = hi[i]; }
  return a;
}

// ---------------------------------------------------------------------------
// Positional-encoding MLP (2->20->64->256, relu) + add descriptors.
// ---------------------------------------------------------------------------
__global__ void pe_kernel(const float* __restrict__ p1, const float* __restrict__ p2,
                          const float* __restrict__ d1, const float* __restrict__ d2,
                          const float* __restrict__ W1, const float* __restrict__ b1,
                          const float* __restrict__ W2, const float* __restrict__ b2,
                          const float* __restrict__ W3, const float* __restrict__ b3,
                          float* __restrict__ x, h16* __restrict__ xh)
{
  int row = blockIdx.x * blockDim.x + threadIdx.x;
  if (row >= NROWS) return;
  const float* pt = (row < NHALF) ? (p1 + (size_t)row * 2) : (p2 + (size_t)(row - NHALF) * 2);
  const float* ds = (row < NHALF) ? (d1 + (size_t)row * EMB) : (d2 + (size_t)(row - NHALF) * EMB);
  float px = pt[0], py = pt[1];
  float ha[20];
#pragma unroll
  for (int j = 0; j < 20; ++j) {
    float v = px * W1[0 * 20 + j] + py * W1[1 * 20 + j] + b1[j];
    ha[j] = fmaxf(v, 0.f);
  }
  float hb[64];
#pragma unroll
  for (int j = 0; j < 64; ++j) {
    float v = b2[j];
#pragma unroll
    for (int k = 0; k < 20; ++k) v += ha[k] * W2[k * 64 + j];
    hb[j] = fmaxf(v, 0.f);
  }
  for (int c = 0; c < EMB; ++c) {
    float v = b3[c];
#pragma unroll
    for (int k = 0; k < 64; ++k) v += hb[k] * W3[k * EMB + c];
    v = fmaxf(v, 0.f);
    float xv = ds[c] + v;
    x[(size_t)row * EMB + c]  = xv;
    xh[(size_t)row * EMB + c] = (h16)xv;
  }
}

// Transpose f32 weight (K x N) -> f16 (N x K) so GEMM B-fragment loads are
// contiguous 32B per lane.
__global__ void transpose_w(const float* __restrict__ W, int K, int N,
                            h16* __restrict__ Wt)
{
  int idx = blockIdx.x * blockDim.x + threadIdx.x;
  if (idx >= K * N) return;
  int k = idx / N, n = idx % N;
  Wt[(size_t)n * K + k] = (h16)W[idx];
}

// ---------------------------------------------------------------------------
// WMMA GEMM: C[M,N] = op(A[M,K] @ Bt[N,K]^T + bias) (+ add). f16 in, f32 acc.
// Compile-time tile: 32 rows x (NT*16) cols per wave, straight-line k-loop
// (2 A-frags reused NT times, each B-frag reused 2 times, 2*NT WMMAs/step).
// Wave w covers n0 = (blockIdx.y*nwaves + w)*(NT*16); callers pick NT and
// waves/block so every wave's tile is fully inside N (no guards needed).
// flags: bit0 = relu, bit1 = add residual (addsrc, same ldc as C).
// Requires M % 32 == 0, K % 32 == 0.
// ---------------------------------------------------------------------------
template<int NT>
__global__ __launch_bounds__(128) void gemm_wmma_t(
    const h16* __restrict__ A, int lda,
    const h16* __restrict__ Bt, int ldb,
    const float* __restrict__ bias,
    const float* __restrict__ addsrc,
    float* __restrict__ C, int ldc,
    h16* __restrict__ Ch, int ldch,
    int M, int N, int K, int flags)
{
  int lane = threadIdx.x & 31;
  int wave = threadIdx.x >> 5;
  int nwaves = blockDim.x >> 5;
  int m0 = blockIdx.x * 32;
  int n0 = (blockIdx.y * nwaves + wave) * (NT * 16);
  int mA = lane & 15, khalf = lane >> 4;
  const h16* arow0 = A + (size_t)(m0 + mA) * lda;
  const h16* arow1 = arow0 + (size_t)16 * lda;
  const h16* brow[NT];
#pragma unroll
  for (int t = 0; t < NT; ++t)
    brow[t] = Bt + (size_t)(n0 + t * 16 + mA) * ldb + khalf * 16;
  v8f acc0[NT] = {};
  v8f acc1[NT] = {};
  for (int k0 = 0; k0 < K; k0 += 32) {
    v16h a0 = load_afrag(arow0, k0, khalf);
    v16h a1 = load_afrag(arow1, k0, khalf);
#pragma unroll
    for (int t = 0; t < NT; ++t) {
      // B fragment (32x16): lane n=mA, slot i <- k = khalf*16 + i (contiguous)
      v16h b = *(const v16h*)(brow[t] + k0);
      acc0[t] = wmma_f16f32(a0, b, acc0[t]);
      acc1[t] = wmma_f16f32(a1, b, acc1[t]);
    }
  }
#pragma unroll
  for (int t = 0; t < NT; ++t) {
    int n = n0 + t * 16 + mA;
    float bn = bias ? bias[n] : 0.f;
#pragma unroll
    for (int r = 0; r < 8; ++r) {
      int m = m0 + r + 8 * khalf;            // C layout: vgpr r, lane group
      float v = acc0[t][r] + bn;
      if (flags & 1) v = fmaxf(v, 0.f);
      if (flags & 2) v += addsrc[(size_t)m * ldc + n];
      C[(size_t)m * ldc + n] = v;
      if (Ch) Ch[(size_t)m * ldch + n] = (h16)v;
      int m2 = m + 16;
      float v2 = acc1[t][r] + bn;
      if (flags & 1) v2 = fmaxf(v2, 0.f);
      if (flags & 2) v2 += addsrc[(size_t)m2 * ldc + n];
      C[(size_t)m2 * ldc + n] = v2;
      if (Ch) Ch[(size_t)m2 * ldch + n] = (h16)v2;
    }
  }
}

// q/k/v scaling by global mins + f16 conversion; V stored transposed (64 x 8192)
// so attention P*V B-fragments load contiguously.
__global__ void qkv_prep(const float* __restrict__ q, const float* __restrict__ kv,
                         const float* __restrict__ scal,
                         h16* __restrict__ qh, h16* __restrict__ kh,
                         h16* __restrict__ vhT)
{
  int idx = blockIdx.x * blockDim.x + threadIdx.x;
  const int NQ = NROWS * KEYD;
  if (idx < NQ) {
    qh[idx] = (h16)(q[idx] / scal[0]);
  } else {
    int j = idx - NQ;
    if (j >= NROWS * (KEYD + VALD)) return;
    int row = j / 96, c = j % 96;
    float v = kv[j];
    if (c < KEYD) kh[(size_t)row * KEYD + c] = (h16)(v / scal[1]);
    else          vhT[(size_t)(c - KEYD) * NROWS + row] = (h16)(v / scal[2]);
  }
}

// ---------------------------------------------------------------------------
// Flash attention: 1 wave = 16 query rows, 32 keys/step.
// S = Q(16x32) x K^T via 2 WMMAs; online softmax; P staged via LDS to
// A-fragment layout; O += P(16x32) x V(32x64) via 4 WMMAs against V^T.
// ---------------------------------------------------------------------------
__global__ __launch_bounds__(128) void attn_kernel(
    const h16* __restrict__ Q, const h16* __restrict__ K,
    const h16* __restrict__ VT, float* __restrict__ O, int nk)
{
  __shared__ __align__(32) h16 plds[4][16][32];
  int lane = threadIdx.x & 31;
  int wave = threadIdx.x >> 5;
  int q0 = (blockIdx.x * 4 + wave) * 16;
  int mA = lane & 15, khalf = lane >> 4;

  v16h aq = load_afrag(Q + (size_t)(q0 + mA) * KEYD, 0, khalf);
  v8f o0 = {}, o1 = {}, o2 = {}, o3 = {};
  float mrun[8], lrun[8];
#pragma unroll
  for (int r = 0; r < 8; ++r) { mrun[r] = -1e30f; lrun[r] = 0.f; }

  for (int kb = 0; kb < nk; kb += 32) {
    v16h bk0 = *(const v16h*)(K + (size_t)(kb + mA) * KEYD + khalf * 16);
    v16h bk1 = *(const v16h*)(K + (size_t)(kb + 16 + mA) * KEYD + khalf * 16);
    v8f z0 = {}, z1 = {};
    v8f s0 = wmma_f16f32(aq, bk0, z0);
    v8f s1 = wmma_f16f32(aq, bk1, z1);
    __syncthreads();
#pragma unroll
    for (int r = 0; r < 8; ++r) {
      float x0 = s0[r], x1 = s1[r];
      float mx = fmaxf(x0, x1);
      mx = fmaxf(mx, __shfl_xor(mx, 1, 16));
      mx = fmaxf(mx, __shfl_xor(mx, 2, 16));
      mx = fmaxf(mx, __shfl_xor(mx, 4, 16));
      mx = fmaxf(mx, __shfl_xor(mx, 8, 16));
      float mnew = fmaxf(mrun[r], mx);
      float corr = __expf(mrun[r] - mnew);
      float p0 = __expf(x0 - mnew), p1 = __expf(x1 - mnew);
      float ps = p0 + p1;
      ps += __shfl_xor(ps, 1, 16);
      ps += __shfl_xor(ps, 2, 16);
      ps += __shfl_xor(ps, 4, 16);
      ps += __shfl_xor(ps, 8, 16);
      lrun[r] = lrun[r] * corr + ps;
      mrun[r] = mnew;
      o0[r] *= corr; o1[r] *= corr; o2[r] *= corr; o3[r] *= corr;
      int m = r + 8 * khalf;                 // C-layout row
      plds[wave][m][mA]      = (h16)p0;      // cols 0..15
      plds[wave][m][16 + mA] = (h16)p1;      // cols 16..31
    }
    __syncthreads();
    v16h ap = load_afrag(&plds[wave][mA][0], 0, khalf);
    const h16* vb = VT + kb + khalf * 16;    // slot i -> key kb + khalf*16 + i
    v16h bv0 = *(const v16h*)(vb + (size_t)(0 * 16 + mA) * NROWS);
    v16h bv1 = *(const v16h*)(vb + (size_t)(1 * 16 + mA) * NROWS);
    v16h bv2 = *(const v16h*)(vb + (size_t)(2 * 16 + mA) * NROWS);
    v16h bv3 = *(const v16h*)(vb + (size_t)(3 * 16 + mA) * NROWS);
    o0 = wmma_f16f32(ap, bv0, o0);
    o1 = wmma_f16f32(ap, bv1, o1);
    o2 = wmma_f16f32(ap, bv2, o2);
    o3 = wmma_f16f32(ap, bv3, o3);
  }
#pragma unroll
  for (int r = 0; r < 8; ++r) {
    int m = r + 8 * khalf;
    float inv = 1.f / lrun[r];
    float* orow = O + (size_t)(q0 + m) * VALD + mA;
    orow[0]  = o0[r] * inv;
    orow[16] = o1[r] * inv;
    orow[32] = o2[r] * inv;
    orow[48] = o3[r] * inv;
  }
}

// cat = [x (f16) | f16(message)]
__global__ void concat_half(const h16* __restrict__ xh, const float* __restrict__ msg,
                            h16* __restrict__ cath)
{
  int idx = blockIdx.x * blockDim.x + threadIdx.x;
  if (idx >= NROWS * CATD) return;
  int row = idx / CATD, c = idx % CATD;
  cath[idx] = (c < EMB) ? xh[(size_t)row * EMB + c]
                        : (h16)msg[(size_t)row * VALD + (c - EMB)];
}

// ------------------------- single-block reductions -------------------------
__global__ void reduce_min(const float* __restrict__ src, int rows, int ld,
                           int c0, int nc, float* __restrict__ out)
{
  __shared__ float sh[256];
  int tid = threadIdx.x;
  float m = 3.4e38f;
  int n = rows * nc;
  for (int i = tid; i < n; i += 256) {
    int r = i / nc, c = c0 + i % nc;
    m = fminf(m, src[(size_t)r * ld + c]);
  }
  sh[tid] = m; __syncthreads();
  for (int s = 128; s > 0; s >>= 1) {
    if (tid < s) sh[tid] = fminf(sh[tid], sh[tid + s]);
    __syncthreads();
  }
  if (tid == 0) *out = sh[0];
}

// mode 0: scale = 1/(0.1*(990 + 0.01*(max-min)))   (per-layer norm)
// mode 1: scale = 1/(990 + 0.01*(max-min))         (final norm)
__global__ void reduce_minmax_scale(const float* __restrict__ src, int n,
                                    float* __restrict__ out, int mode)
{
  __shared__ float smin[256], smax[256];
  int tid = threadIdx.x;
  float lo = 3.4e38f, hi = -3.4e38f;
  for (int i = tid; i < n; i += 256) {
    float v = src[i];
    lo = fminf(lo, v); hi = fmaxf(hi, v);
  }
  smin[tid] = lo; smax[tid] = hi; __syncthreads();
  for (int s = 128; s > 0; s >>= 1) {
    if (tid < s) {
      smin[tid] = fminf(smin[tid], smin[tid + s]);
      smax[tid] = fmaxf(smax[tid], smax[tid + s]);
    }
    __syncthreads();
  }
  if (tid == 0) {
    float norm = 990.f + 0.01f * (smax[0] - smin[0]);
    *out = (mode == 0) ? 1.f / (0.1f * norm) : 1.f / norm;
  }
}

__global__ void reduce_sum_inv(const float* __restrict__ src, long long n,
                               float* __restrict__ out)
{
  __shared__ float sh[256];
  int tid = threadIdx.x;
  float s = 0.f;
  for (long long i = tid; i < n; i += 256) s += src[i];
  sh[tid] = s; __syncthreads();
  for (int st = 128; st > 0; st >>= 1) {
    if (tid < st) sh[tid] += sh[tid + st];
    __syncthreads();
  }
  if (tid == 0) *out = 1.f / sh[0];
}

__global__ void scale_to_half(float* __restrict__ src, h16* __restrict__ dst,
                              int n, const float* __restrict__ sp)
{
  int i = blockIdx.x * blockDim.x + threadIdx.x;
  if (i >= n) return;
  float v = src[i] * sp[0];
  src[i] = v;
  dst[i] = (h16)v;
}

__global__ void scale_inplace(float* __restrict__ P, long long n,
                              const float* __restrict__ sp)
{
  long long i = (long long)blockIdx.x * blockDim.x + threadIdx.x;
  if (i < n) P[i] *= sp[0];
}

// --------------------------------- Sinkhorn --------------------------------
__global__ void fill_bins(float* __restrict__ S, const float* __restrict__ binp)
{
  int idx = blockIdx.x * blockDim.x + threadIdx.x;
  float b = binp[0];
  if (idx < NP)                 S[(size_t)NHALF * NP + idx] = b;        // last row
  else if (idx < NP + NHALF)    S[(size_t)(idx - NP) * NP + NHALF] = b; // last col
}

__global__ void exp_init(const float* __restrict__ S, float* __restrict__ P,
                         long long n)
{
  long long i = (long long)blockIdx.x * blockDim.x + threadIdx.x;
  if (i < n) P[i] = __expf(0.3f * S[i]);
}

__global__ void rownorm(float* __restrict__ P)
{
  __shared__ float sh[256];
  __shared__ float fac;
  int i = blockIdx.x, tid = threadIdx.x;
  float s = 0.f;
  for (int j = tid; j < NP; j += 256) s += P[(size_t)i * NP + j];
  sh[tid] = s; __syncthreads();
  for (int st = 128; st > 0; st >>= 1) {
    if (tid < st) sh[tid] += sh[tid + st];
    __syncthreads();
  }
  if (tid == 0) {
    float r = (i == NHALF) ? (float)NHALF : 1.f;
    fac = r / sh[0];
  }
  __syncthreads();
  float f = fac;
  for (int j = tid; j < NP; j += 256) P[(size_t)i * NP + j] *= f;
}

__global__ void colnorm(float* __restrict__ P)
{
  int j = blockIdx.x * blockDim.x + threadIdx.x;
  if (j >= NP) return;
  float s = 0.f;
  for (int i = 0; i < NP; ++i) s += P[(size_t)i * NP + j];
  float c = (j == NHALF) ? (float)NHALF : 1.f;
  float f = c / s;
  for (int i = 0; i < NP; ++i) P[(size_t)i * NP + j] *= f;
}

__global__ void cost_kernel(const float* __restrict__ P, const float* __restrict__ S,
                            long long n, float* __restrict__ out)
{
  __shared__ float sh[256];
  int tid = threadIdx.x;
  float s = 0.f;
  for (long long i = tid; i < n; i += 256) s += P[i] * S[i];
  sh[tid] = s; __syncthreads();
  for (int st = 128; st > 0; st >>= 1) {
    if (tid < st) sh[tid] += sh[tid + st];
    __syncthreads();
  }
  if (tid == 0) *out = -sh[0];   // cost = sum(P * M), M = -S
}

// ---------------------------------------------------------------------------
extern "C" void kernel_launch(void* const* d_in, const int* in_sizes, int n_in,
                              void* d_out, int out_size, void* d_ws, size_t ws_size,
                              hipStream_t stream)
{
  (void)n_in; (void)out_size; (void)ws_size;
  const float* points1 = (const float*)d_in[0];
  const float* points2 = (const float*)d_in[1];
  const float* desc1   = (const float*)d_in[2];
  const float* desc2   = (const float*)d_in[3];

  const float *pe1W,*pe1b,*pe2W,*pe2b,*pe3W,*pe3b,*projW,*projb,*qprojW,*qprojb;
  const float *mm1W,*mm1b,*mm2W,*mm2b,*mm3W,*mm3b,*finW,*finb,*binp;
  if (in_sizes[4] == 1) {   // sorted-key flatten (jax pytree order)
    binp  = (const float*)d_in[4];
    finW  = (const float*)d_in[5];  finb  = (const float*)d_in[6];
    mm1W  = (const float*)d_in[7];  mm1b  = (const float*)d_in[8];
    mm2W  = (const float*)d_in[9];  mm2b  = (const float*)d_in[10];
    mm3W  = (const float*)d_in[11]; mm3b  = (const float*)d_in[12];
    pe1W  = (const float*)d_in[13]; pe1b  = (const float*)d_in[14];
    pe2W  = (const float*)d_in[15]; pe2b  = (const float*)d_in[16];
    pe3W  = (const float*)d_in[17]; pe3b  = (const float*)d_in[18];
    projW = (const float*)d_in[19]; projb = (const float*)d_in[20];
    qprojW= (const float*)d_in[21]; qprojb= (const float*)d_in[22];
  } else {                  // insertion order
    pe1W  = (const float*)d_in[4];  pe1b  = (const float*)d_in[5];
    pe2W  = (const float*)d_in[6];  pe2b  = (const float*)d_in[7];
    pe3W  = (const float*)d_in[8];  pe3b  = (const float*)d_in[9];
    projW = (const float*)d_in[10]; projb = (const float*)d_in[11];
    qprojW= (const float*)d_in[12]; qprojb= (const float*)d_in[13];
    mm1W  = (const float*)d_in[14]; mm1b  = (const float*)d_in[15];
    mm2W  = (const float*)d_in[16]; mm2b  = (const float*)d_in[17];
    mm3W  = (const float*)d_in[18]; mm3b  = (const float*)d_in[19];
    finW  = (const float*)d_in[20]; finb  = (const float*)d_in[21];
    binp  = (const float*)d_in[22];
  }

  // ---- workspace carve (floats, 256B aligned chunks) ----
  float* wsf = (float*)d_ws;
  size_t off = 0;
  auto alloc_f = [&](size_t nf) -> float* {
    float* p = wsf + off; off += (nf + 63) & ~(size_t)63; return p;
  };
  auto alloc_h = [&](size_t nh) -> h16* { return (h16*)alloc_f((nh + 1) / 2); };

  float* scal = alloc_f(64);
  float* x    = alloc_f((size_t)NROWS * EMB);
  h16*   xh   = alloc_h((size_t)NROWS * EMB);
  float* q    = alloc_f((size_t)NROWS * KEYD);
  float* kv   = alloc_f((size_t)NROWS * 96);
  h16*   qh   = alloc_h((size_t)NROWS * KEYD);
  h16*   kh   = alloc_h((size_t)NROWS * KEYD);
  h16*   vhT  = alloc_h((size_t)VALD * NROWS);
  float* msg  = alloc_f((size_t)NROWS * VALD);
  h16*   cath = alloc_h((size_t)NROWS * CATD);
  float* h1   = alloc_f((size_t)NROWS * EMB);
  h16*   h1h  = alloc_h((size_t)NROWS * EMB);
  float* h2   = alloc_f((size_t)NROWS * EMB);
  h16*   h2h  = alloc_h((size_t)NROWS * EMB);
  float* fbuf = alloc_f((size_t)NROWS * EMB);
  h16*   fh   = alloc_h((size_t)NROWS * EMB);
  h16*   qprojT = alloc_h((size_t)EMB * KEYD);
  h16*   projT  = alloc_h((size_t)EMB * 96);
  h16*   mm1T   = alloc_h((size_t)CATD * EMB);
  h16*   mm2T   = alloc_h((size_t)EMB * EMB);
  h16*   mm3T   = alloc_h((size_t)EMB * EMB);
  h16*   finT   = alloc_h((size_t)EMB * EMB);
  float* S      = alloc_f((size_t)NP * NP);

  float* P = (float*)d_out;
  float* costp = P + NPSQ;

  // ---- prep: weights -> f16 transposed ----
  transpose_w<<<(EMB * KEYD + 255) / 256, 256, 0, stream>>>(qprojW, EMB, KEYD, qprojT);
  transpose_w<<<(EMB * 96 + 255) / 256, 256, 0, stream>>>(projW, EMB, 96, projT);
  transpose_w<<<(CATD * EMB + 255) / 256, 256, 0, stream>>>(mm1W, CATD, EMB, mm1T);
  transpose_w<<<(EMB * EMB + 255) / 256, 256, 0, stream>>>(mm2W, EMB, EMB, mm2T);
  transpose_w<<<(EMB * EMB + 255) / 256, 256, 0, stream>>>(mm3W, EMB, EMB, mm3T);
  transpose_w<<<(EMB * EMB + 255) / 256, 256, 0, stream>>>(finW, EMB, EMB, finT);

  pe_kernel<<<32, 256, 0, stream>>>(points1, points2, desc1, desc2,
                                    pe1W, pe1b, pe2W, pe2b, pe3W, pe3b, x, xh);

  const int NE = NROWS * EMB;
  for (int layer = 0; layer < 5; ++layer) {
    // q: N=32 -> NT=2, 1 wave/block covers all 32 cols
    gemm_wmma_t<2><<<dim3(NROWS / 32, 1), 32, 0, stream>>>(
        xh, EMB, qprojT, EMB, qprojb, nullptr, q, KEYD, nullptr, 0,
        NROWS, KEYD, EMB, 0);
    // kv: N=96 -> NT=2, 3 waves/block (32 cols each)
    gemm_wmma_t<2><<<dim3(NROWS / 32, 1), 96, 0, stream>>>(
        xh, EMB, projT, EMB, projb, nullptr, kv, 96, nullptr, 0,
        NROWS, 96, EMB, 0);
    reduce_min<<<1, 256, 0, stream>>>(q, NROWS, KEYD, 0, KEYD, scal + 0);
    reduce_min<<<1, 256, 0, stream>>>(kv, NROWS, 96, 0, 32, scal + 1);
    reduce_min<<<1, 256, 0, stream>>>(kv, NROWS, 96, 32, 64, scal + 2);
    qkv_prep<<<(NROWS * 128 + 255) / 256, 256, 0, stream>>>(q, kv, scal, qh, kh, vhT);

    int cross = layer & 1;
    // m_s : queries 0..4095 attend keys of (cross ? target : source)
    attn_kernel<<<NHALF / 64, 128, 0, stream>>>(
        qh, kh + (cross ? (size_t)NHALF * KEYD : 0),
        vhT + (cross ? NHALF : 0), msg, NHALF);
    // m_t : queries 4096..8191 attend keys of (cross ? source : target)
    attn_kernel<<<NHALF / 64, 128, 0, stream>>>(
        qh + (size_t)NHALF * KEYD, kh + (cross ? 0 : (size_t)NHALF * KEYD),
        vhT + (cross ? 0 : NHALF), msg + (size_t)NHALF * VALD, NHALF);

    concat_half<<<(NROWS * CATD + 255) / 256, 256, 0, stream>>>(xh, msg, cath);
    // MLP GEMMs: N=256 -> NT=4, 4 waves/block (64 cols each)
    gemm_wmma_t<4><<<dim3(NROWS / 32, 1), 128, 0, stream>>>(
        cath, CATD, mm1T, CATD, mm1b, nullptr, h1, EMB, h1h, EMB,
        NROWS, EMB, CATD, 1);
    gemm_wmma_t<4><<<dim3(NROWS / 32, 1), 128, 0, stream>>>(
        h1h, EMB, mm2T, EMB, mm2b, nullptr, h2, EMB, h2h, EMB,
        NROWS, EMB, EMB, 1);
    gemm_wmma_t<4><<<dim3(NROWS / 32, 1), 128, 0, stream>>>(
        h2h, EMB, mm3T, EMB, mm3b, x, x, EMB, nullptr, 0,
        NROWS, EMB, EMB, 3);
    reduce_minmax_scale<<<1, 256, 0, stream>>>(x, NE, scal + 3, 0);
    scale_to_half<<<(NE + 255) / 256, 256, 0, stream>>>(x, xh, NE, scal + 3);
  }

  // ---- final projection + norm ----
  gemm_wmma_t<4><<<dim3(NROWS / 32, 1), 128, 0, stream>>>(
      xh, EMB, finT, EMB, finb, nullptr, fbuf, EMB, nullptr, 0,
      NROWS, EMB, EMB, 0);
  reduce_minmax_scale<<<1, 256, 0, stream>>>(fbuf, NE, scal + 4, 1);
  scale_to_half<<<(NE + 255) / 256, 256, 0, stream>>>(fbuf, fh, NE, scal + 4);

  // ---- score = f1 @ f2^T into S (ld = 4097), bins, Sinkhorn ----
  gemm_wmma_t<4><<<dim3(NHALF / 32, NHALF / 256), 128, 0, stream>>>(
      fh, EMB, fh + (size_t)NHALF * EMB, EMB, nullptr, nullptr, S, NP,
      nullptr, 0, NHALF, NHALF, EMB, 0);
  fill_bins<<<(NP + NHALF + 255) / 256, 256, 0, stream>>>(S, binp);

  exp_init<<<(int)((NPSQ + 255) / 256), 256, 0, stream>>>(S, P, NPSQ);
  reduce_sum_inv<<<1, 256, 0, stream>>>(P, NPSQ, scal + 5);
  scale_inplace<<<(int)((NPSQ + 255) / 256), 256, 0, stream>>>(P, NPSQ, scal + 5);
  for (int it = 0; it < 100; ++it) {
    rownorm<<<NP, 256, 0, stream>>>(P);
    colnorm<<<(NP + 255) / 256, 256, 0, stream>>>(P);
  }
  cost_kernel<<<1, 256, 0, stream>>>(P, S, NPSQ, costp);
}
